// TimeThenSpaceModel_64321430225491
// MI455X (gfx1250) — matrix operand order repaired and checked by reading, not compiled, not run on hardware
//
#include <hip/hip_runtime.h>
#include <math.h>

// ---- problem constants -----------------------------------------------------
constexpr int B_   = 8;
constexpr int T_   = 12;
constexpr int N_   = 10000;
constexpr int H_   = 32;
constexpr int E_   = 320000;
constexpr int HOR_ = 12;
constexpr int ROWS_ = B_ * N_;          // 80000, divisible by 16
constexpr int TILES_ = ROWS_ / 16;      // 5000
constexpr int WAVES_PER_BLOCK = 8;      // 256 threads = 8 wave32

typedef __attribute__((ext_vector_type(16))) _Float16 v16h;
typedef __attribute__((ext_vector_type(8)))  float    v8f;

__device__ __forceinline__ v8f wmma_f16(v16h a, v16h b, v8f c) {
  // D = A(16x32 f16) * B(32x16 f16) + C(16x16 f32)
  return __builtin_amdgcn_wmma_f32_16x16x32_f16(
      /*neg_a=*/false, a, /*neg_b=*/false, b,
      /*c_mod=*/(short)0, c, /*reuse_a=*/false, /*reuse_b=*/false);
}

// fast transcendentals: single v_exp_f32 + single v_rcp_f32 each,
// avoiding the IEEE div_scale/div_fmas Newton sequence.
__device__ __forceinline__ float fast_sigmoid(float x) {
  return __builtin_amdgcn_rcpf(1.0f + __expf(-x));
}
__device__ __forceinline__ float fast_tanh(float x) {
  // tanh(x) = 1 - 2/(exp(2x)+1); saturates to +/-1 correctly at both ends
  return 1.0f - 2.0f * __builtin_amdgcn_rcpf(__expf(2.0f * x) + 1.0f);
}

// ---------------------------------------------------------------------------
// zero helper
__global__ void zero_f32(float* __restrict__ p, int count) {
  int i = blockIdx.x * blockDim.x + threadIdx.x;
  if (i < count) p[i] = 0.0f;
}

// ---------------------------------------------------------------------------
// Fused encoder + node-emb + 12-step GRU. One wave32 per 16-row tile.
// A-matrix layout (16-bit, 16x32): lane<16 -> M=lane, K={kh..kh+7, kh+16..kh+23}, kh=0
//                                  lane>=16 -> M=lane-16, kh=8
// B-matrix layout (32x16 f16):     N = lane%16, elt j = same K as A's elt j
// C/D layout (16x16 f32):          lane<16: vgpr v = D[v][lane]; lane>=16: D[v+8][lane-16]
__global__ __launch_bounds__(256) void gru_kernel(
    const float* __restrict__ x,        // [B,T,N,1]
    const float* __restrict__ enc_w,    // [1,H]
    const float* __restrict__ enc_b,    // [H]
    const float* __restrict__ node_emb, // [N,H]
    const float* __restrict__ w_ih,     // [3H,H]
    const float* __restrict__ w_hh,     // [3H,H]
    const float* __restrict__ b_ih,     // [3H]
    const float* __restrict__ b_hh,     // [3H]
    float* __restrict__ h_time)         // [ROWS,H]
{
  __shared__ float hlds_all[WAVES_PER_BLOCK][16 * H_];
  const int lane = threadIdx.x & 31;
  const int wave = threadIdx.x >> 5;
  const int tile = blockIdx.x * WAVES_PER_BLOCK + wave;
  if (tile >= TILES_) return;                       // wave-uniform branch
  float* hlds = hlds_all[wave];

  const int m    = lane & 15;                       // row within tile (A) / column (D)
  const int row  = tile * 16 + m;
  const int b    = row / N_;
  const int n    = row % N_;
  const int kh   = (lane < 16) ? 0 : 8;
  const int mrow = (lane < 16) ? 0 : 8;             // D-layout base row

  // per-lane encoder coefficients for this lane's 16 K slots
  float ew[16], basev[16];
#pragma unroll
  for (int i = 0; i < 16; ++i) {
    int k = (i < 8) ? (kh + i) : (kh + 8 + i);      // kh+i  or  kh+16+(i-8)
    ew[i]    = enc_w[k];
    basev[i] = enc_b[k] + node_emb[n * H_ + k];
  }

  // resident B fragments: gate tiles r(0,1) z(2,3) n(4,5)
  v16h Bih[6], Bhh[6];
#pragma unroll
  for (int gt = 0; gt < 6; ++gt) {
    int g0 = gt * 16;
    v16h bi, bh;
#pragma unroll
    for (int j = 0; j < 16; ++j) {
      bi[j] = (_Float16)w_ih[(g0 + m) * H_ + ((j < 8) ? (kh + j) : (kh + 8 + j))]; // B[k][n=m] = w_ih[g][k]
      bh[j] = (_Float16)w_hh[(g0 + m) * H_ + ((j < 8) ? (kh + j) : (kh + 8 + j))];
    }
    Bih[gt] = bi; Bhh[gt] = bh;
  }

  // per-lane gate biases (column f = ft*16 + m)
  float bx[6], bh_[6];
#pragma unroll
  for (int ft = 0; ft < 2; ++ft) {
#pragma unroll
    for (int g = 0; g < 3; ++g) {
      bx[g * 2 + ft]  = b_ih[g * H_ + ft * 16 + m];
      bh_[g * 2 + ft] = b_hh[g * H_ + ft * 16 + m];
    }
  }

  // h = 0
  for (int i = lane; i < 16 * H_; i += 32) hlds[i] = 0.0f;
  __builtin_amdgcn_wave_barrier();

  for (int t = 0; t < T_; ++t) {
    // x_emb A-fragment, built in registers (F_IN == 1)
    const float xv = x[(b * T_ + t) * N_ + n];
    v16h aX, aH;
#pragma unroll
    for (int i = 0; i < 16; ++i) aX[i] = (_Float16)(xv * ew[i] + basev[i]);
    // h A-fragment from LDS (previous step's state)
#pragma unroll
    for (int i = 0; i < 8; ++i) {
      aH[i]     = (_Float16)hlds[m * H_ + kh + i];
      aH[i + 8] = (_Float16)hlds[m * H_ + kh + 16 + i];
    }
    __builtin_amdgcn_wave_barrier();

#pragma unroll
    for (int ft = 0; ft < 2; ++ft) {                // feature column tiles 0..15 / 16..31
      v8f xr = {}, xz = {}, xn = {}, hr = {}, hz = {}, hn = {};
      xr = wmma_f16(aX, Bih[0 + ft], xr);
      xz = wmma_f16(aX, Bih[2 + ft], xz);
      xn = wmma_f16(aX, Bih[4 + ft], xn);
      hr = wmma_f16(aH, Bhh[0 + ft], hr);
      hz = wmma_f16(aH, Bhh[2 + ft], hz);
      hn = wmma_f16(aH, Bhh[4 + ft], hn);

      const int f = ft * 16 + m;                    // feature column
      const float bxr = bx[0 + ft], bhr = bh_[0 + ft];
      const float bxz = bx[2 + ft], bhz = bh_[2 + ft];
      const float bxn = bx[4 + ft], bhn = bh_[4 + ft];

      float hnew[8];
#pragma unroll
      for (int v = 0; v < 8; ++v) {
        float hold = hlds[(mrow + v) * H_ + f];
        float r  = fast_sigmoid(xr[v] + bxr + hr[v] + bhr);
        float z  = fast_sigmoid(xz[v] + bxz + hz[v] + bhz);
        float nn = fast_tanh(xn[v] + bxn + r * (hn[v] + bhn));
        hnew[v]  = (1.0f - z) * nn + z * hold;
      }
#pragma unroll
      for (int v = 0; v < 8; ++v) hlds[(mrow + v) * H_ + f] = hnew[v];
    }
    __builtin_amdgcn_wave_barrier();
  }

  // write last hidden state
  for (int i = lane; i < 16 * H_; i += 32) h_time[tile * (16 * H_) + i] = hlds[i];
}

// ---------------------------------------------------------------------------
// degree accumulation and edge-weight normalization
__global__ void degree_kernel(const int* __restrict__ src, const int* __restrict__ tgt,
                              const float* __restrict__ ew,
                              float* __restrict__ deg_f, float* __restrict__ deg_b) {
  int e = blockIdx.x * blockDim.x + threadIdx.x;
  if (e >= E_) return;
  atomicAdd(&deg_f[tgt[e]], ew[e]);
  atomicAdd(&deg_b[src[e]], ew[e]);
}

__global__ void norm_kernel(const int* __restrict__ src, const int* __restrict__ tgt,
                            const float* __restrict__ ew,
                            const float* __restrict__ deg_f, const float* __restrict__ deg_b,
                            float* __restrict__ w_f, float* __restrict__ w_b) {
  int e = blockIdx.x * blockDim.x + threadIdx.x;
  if (e >= E_) return;
  float df = deg_f[tgt[e]];
  float db = deg_b[src[e]];
  w_f[e] = ew[e] / (df > 0.0f ? df : 1.0f);
  w_b[e] = ew[e] / (db > 0.0f ? db : 1.0f);
}

// ---------------------------------------------------------------------------
// one diffusion hop: out[b, dst[e], :] += w[e] * xin[b, sidx[e], :]
// one thread per (edge, batch)
__global__ void prop_kernel(const float* __restrict__ xin,
                            const int* __restrict__ sidx, const int* __restrict__ didx,
                            const float* __restrict__ wn,
                            float* __restrict__ xout) {
  int gid = blockIdx.x * blockDim.x + threadIdx.x;
  int e  = gid >> 3;
  int bb = gid & 7;
  if (e >= E_) return;
  const int s = sidx[e], d = didx[e];
  const float w = wn[e];
  const float* __restrict__ xi = xin  + ((size_t)bb * N_ + s) * H_;
  float*       __restrict__ xo = xout + ((size_t)bb * N_ + d) * H_;
  float vals[H_];
#pragma unroll
  for (int k = 0; k < H_; ++k) vals[k] = xi[k];
#pragma unroll
  for (int k = 0; k < H_; ++k) atomicAdd(&xo[k], w * vals[k]);
}

// ---------------------------------------------------------------------------
// filter ([rows,160] @ [160,32] + b) then decoder ([rows,32] @ [32,12] + b),
// writing out[b, t, n, 0]. One wave32 per 16-row tile.
__global__ __launch_bounds__(256) void filt_dec_kernel(
    const float* __restrict__ s0,  // h_time
    const float* __restrict__ s1,  // hop f1
    const float* __restrict__ s2,  // hop f2
    const float* __restrict__ s3,  // hop b1
    const float* __restrict__ s4,  // hop b2
    const float* __restrict__ filt_w,  // [160,32]
    const float* __restrict__ filt_b,  // [32]
    const float* __restrict__ dec_w,   // [32,12]
    const float* __restrict__ dec_b,   // [12]
    float* __restrict__ out)           // [B,12,N,1]
{
  __shared__ float zlds_all[WAVES_PER_BLOCK][16 * H_];
  const int lane = threadIdx.x & 31;
  const int wave = threadIdx.x >> 5;
  const int tile = blockIdx.x * WAVES_PER_BLOCK + wave;
  if (tile >= TILES_) return;
  float* zlds = zlds_all[wave];

  const int m    = lane & 15;
  const int kh   = (lane < 16) ? 0 : 8;
  const int mrow = (lane < 16) ? 0 : 8;

  // A fragments for the 5 concatenated K-tiles, straight from global
  const float* srcs[5] = { s0, s1, s2, s3, s4 };
  v16h aS[5];
#pragma unroll
  for (int kt = 0; kt < 5; ++kt) {
    const float* sp = srcs[kt] + (size_t)(tile * 16 + m) * H_;
    v16h a;
#pragma unroll
    for (int i = 0; i < 8; ++i) {
      a[i]     = (_Float16)sp[kh + i];
      a[i + 8] = (_Float16)sp[kh + 16 + i];
    }
    aS[kt] = a;
  }

  // z = cat @ filt_w + filt_b   (B used untransposed: B[k][j] = filt_w[k][j])
#pragma unroll
  for (int nt = 0; nt < 2; ++nt) {
    v8f c = {};
#pragma unroll
    for (int kt = 0; kt < 5; ++kt) {
      v16h bF;
#pragma unroll
      for (int j = 0; j < 16; ++j)
        bF[j] = (_Float16)filt_w[(kt * H_ + ((j < 8) ? (kh + j) : (kh + 8 + j))) * H_ + nt * 16 + m];
      c = wmma_f16(aS[kt], bF, c);
    }
    const float fb = filt_b[nt * 16 + m];
#pragma unroll
    for (int v = 0; v < 8; ++v)
      zlds[(mrow + v) * H_ + nt * 16 + m] = c[v] + fb;  // store z in D layout
  }
  __builtin_amdgcn_wave_barrier();

  // decoder: A = z (from LDS), B = dec_w padded 12->16 columns
  v16h aZ, bD;
#pragma unroll
  for (int i = 0; i < 8; ++i) {
    aZ[i]     = (_Float16)zlds[m * H_ + kh + i];
    aZ[i + 8] = (_Float16)zlds[m * H_ + kh + 16 + i];
  }
#pragma unroll
  for (int j = 0; j < 16; ++j)
    bD[j] = (m < HOR_) ? (_Float16)dec_w[((j < 8) ? (kh + j) : (kh + 8 + j)) * HOR_ + m] : (_Float16)0.0f;

  v8f d = {};
  d = wmma_f16(aZ, bD, d);

  // scatter: out[b, t=m, n, 0]   (columns m >= 12 are padding)
  if (m < HOR_) {
    const float db = dec_b[m];
#pragma unroll
    for (int v = 0; v < 8; ++v) {
      int grow = tile * 16 + mrow + v;
      int bb = grow / N_;
      int nn = grow % N_;
      out[((size_t)bb * HOR_ + m) * N_ + nn] = d[v] + db;
    }
  }
}

// ---------------------------------------------------------------------------
extern "C" void kernel_launch(void* const* d_in, const int* in_sizes, int n_in,
                              void* d_out, int out_size, void* d_ws, size_t ws_size,
                              hipStream_t stream) {
  const float* x        = (const float*)d_in[0];
  const int*   eidx     = (const int*)d_in[1];   // [2,E] int32 (JAX x64 off)
  const float* ew       = (const float*)d_in[2];
  const float* enc_w    = (const float*)d_in[3];
  const float* enc_b    = (const float*)d_in[4];
  const float* node_emb = (const float*)d_in[5];
  const float* w_ih     = (const float*)d_in[6];
  const float* w_hh     = (const float*)d_in[7];
  const float* b_ih     = (const float*)d_in[8];
  const float* b_hh     = (const float*)d_in[9];
  const float* filt_w   = (const float*)d_in[10];
  const float* filt_b   = (const float*)d_in[11];
  const float* dec_w    = (const float*)d_in[12];
  const float* dec_b    = (const float*)d_in[13];
  float* out = (float*)d_out;
  float* ws  = (float*)d_ws;

  const int* src = eidx;         // edge_index[0]
  const int* tgt = eidx + E_;    // edge_index[1]

  // workspace layout (floats)
  constexpr int NFEAT = ROWS_ * H_;              // 2,560,000
  float* h_time = ws;
  float* sf1    = ws + 1 * NFEAT;
  float* sf2    = ws + 2 * NFEAT;
  float* sb1    = ws + 3 * NFEAT;
  float* sb2    = ws + 4 * NFEAT;
  float* deg_f  = ws + 5 * NFEAT;
  float* deg_b  = deg_f + N_;
  float* w_f    = deg_b + N_;
  float* w_b    = w_f + E_;

  // zero everything written by atomics (sf1..sb2, deg_f, deg_b)
  {
    int cnt = 4 * NFEAT + 2 * N_;
    zero_f32<<<(cnt + 255) / 256, 256, 0, stream>>>(sf1, cnt);
  }

  // fused encoder + GRU
  gru_kernel<<<TILES_ / WAVES_PER_BLOCK, 256, 0, stream>>>(
      x, enc_w, enc_b, node_emb, w_ih, w_hh, b_ih, b_hh, h_time);

  // graph normalization
  degree_kernel<<<(E_ + 255) / 256, 256, 0, stream>>>(src, tgt, ew, deg_f, deg_b);
  norm_kernel<<<(E_ + 255) / 256, 256, 0, stream>>>(src, tgt, ew, deg_f, deg_b, w_f, w_b);

  // diffusion hops: forward (src->tgt, w_f), backward (tgt->src, w_b)
  const int pgrid = (E_ * 8 + 255) / 256;
  prop_kernel<<<pgrid, 256, 0, stream>>>(h_time, src, tgt, w_f, sf1);
  prop_kernel<<<pgrid, 256, 0, stream>>>(sf1,    src, tgt, w_f, sf2);
  prop_kernel<<<pgrid, 256, 0, stream>>>(h_time, tgt, src, w_b, sb1);
  prop_kernel<<<pgrid, 256, 0, stream>>>(sb1,    tgt, src, w_b, sb2);

  // filter + decoder + output transpose
  filt_dec_kernel<<<TILES_ / WAVES_PER_BLOCK, 256, 0, stream>>>(
      h_time, sf1, sf2, sb1, sb2, filt_w, filt_b, dec_w, dec_b, out);
}
